// PlotterEmbedding_19902878450186
// MI455X (gfx1250) — matrix operand order repaired and checked by reading, not compiled, not run on hardware
//
#include <hip/hip_runtime.h>

// ---------------------------------------------------------------------------
// PlotterEmbedding on MI455X (gfx1250, wave32, WMMA).
//
// emb = A(65536x600, band-scattered Gaussians) x W^T(600x64) + bias.
// A is built by SCATTER: each lane computes only its row's 50 nonzero
// Gaussian values (structural minimum) and ds_store_b16's them into a
// wave-private LDS row; the K-loop reloads them as two contiguous
// ds_load_b128 per k-step in the exact wave32 WMMA A layout. W^T is staged
// once per block into LDS pre-swizzled into the WMMA B layout.
// After use, the lane re-zeroes exactly the slots it wrote.
// 8 waves/block (2 waves per SIMD32) to hide ds_load->wmma wait chains;
// 242 KB LDS fits the 320 KB WGP budget.
// Memory-bound target: 16.7 MB fp32 out ~ 0.7 us @ 23.3 TB/s.
// ---------------------------------------------------------------------------

typedef __attribute__((ext_vector_type(16))) _Float16 v16h;
typedef __attribute__((ext_vector_type(8)))  _Float16 v8h;
typedef __attribute__((ext_vector_type(8)))  float    v8f;

#define TDIM     1024
#define HDIM     100
#define NBANDS   6
#define ODIM     64
#define KDIM     600
#define KSTEPS   20            // K padded to 640
#define NT4      4             // ODIM/16
#define AROWH    648           // padded A row stride in halves (1296 B -> bank spread)
#define NROWTILES 4096         // 65536 rows / 16
#define NTHREADS 256
#define WAVES    8
#define NBLOCKS  256
#define B_HALVES (KSTEPS * NT4 * 32 * 16)   // 40960 halves = 80 KB
#define A_HALVES (WAVES * 16 * AROWH)       // 82944 halves = 162 KB

extern __shared__ v16h smemVec[];   // B region then A region; 242 KB total

__global__ __launch_bounds__(NTHREADS) void plotter_emb_wmma(
    const float* __restrict__ tin,  const float* __restrict__ fin,
    const int*   __restrict__ band, const int*   __restrict__ msk,
    const float* __restrict__ W,    const float* __restrict__ bias,
    float* __restrict__ out)
{
    _Float16* sB = (_Float16*)smemVec;
    _Float16* sA = sB + B_HALVES;

    const int tid  = threadIdx.x;
    const int wave = tid >> 5;          // 8 waves / block (wave32)
    const int lane = tid & 31;
    const int mrow = lane & 15;         // row within 16-row tile owned by this lane
    const int hi16 = (lane >> 4) & 1;   // lane >= 16 half
    const int h0   = hi16 * 50;         // this lane scatters h in [h0, h0+50)

    // ---- zero B pad + A region (value slots are restored after every tile) ----
    {
        v16h z = {};
        for (int i = tid; i < (B_HALVES + A_HALVES) / 16; i += NTHREADS) smemVec[i] = z;
    }
    __syncthreads();

    // ---- stage W^T as f16 into LDS, pre-swizzled into WMMA B layout ----
    // B tile (ks,nt): lane l reads 16 contiguous halves at
//   sB[((ks*4+nt)*32 + l)*16 + j] == W^T[ks*32 + ((l&16)>>1) + (j&7) + ((j&8)<<1)][nt*16 + (l&15)]
    for (int idx = tid; idx < ODIM * KDIM; idx += NTHREADS) {
        int n  = idx / KDIM;            // coalesced read of W[n][k]
        int k  = idx - n * KDIM;
        int ks = k >> 5, kin = k & 31;
        int nt = n >> 4, nl  = n & 15;
        sB[(((ks << 2) + nt) * 32 + (nl + ((kin & 8) << 1))) * 16 + ((kin & 7) | ((kin & 16) >> 1))] = (_Float16)W[idx];
    }
    __syncthreads();

    const float coef = 0.3989422804014327f;            // 1/sqrt(2*pi)
    _Float16* aRow = sA + (wave * 16 + mrow) * AROWH;  // shared by lanes l and l+16

    // each wave processes NROWTILES / (NBLOCKS*WAVES) = 2 row tiles
    for (int tile = blockIdx.x * WAVES + wave; tile < NROWTILES; tile += NBLOCKS * WAVES) {
        const int base = tile << 4;
        const int row  = base + mrow;   // lanes 16-31 duplicate lanes 0-15

        // ---- per-row scalars ----
        float tv = tin[row];
        float fv = fin[row];
        int   bd = band[row];
        int   mk = msk[row];
        float tmap = (1.0f / (1.0f + __expf(-tv)) - 0.5f) * 2.0f * (float)TDIM;
        float fmap = (1.0f / (1.0f + __expf(-fv))) * (float)HDIM;
        if (!mk) { tmap = -10.0f; fmap = -10.0f; }  // fill: Gaussians underflow to 0
        float dx = (float)(row & (TDIM - 1)) - tmap;
        float s  = coef * __expf(-0.5f * dx * dx);  // per-row time factor folded into A

        // ---- phase 1: scatter this lane's 50 Gaussian values into the A row ----
        // slot k = 6h + band  (always < 600); everything else is already zero
        {
            _Float16* ap = aRow + bd + 6 * h0;
            for (int i = 0; i < 50; ++i) {
                float dy = (float)(h0 + i) - fmap;
                ap[6 * i] = (_Float16)(s * __expf(-0.5f * dy * dy));
            }
        }

        // ---- phase 2: K loop, pure ds_load + wmma (LDS ops in-order per wave) ----
        v8f c[NT4] = {};
        for (int ks = 0; ks < KSTEPS; ++ks) {
            const _Float16* ap = aRow + (ks << 5) + (hi16 << 3);
            v8h alo = *(const v8h*)ap;          // K = ks*32 + hi*8 + 0..7
            v8h ahi = *(const v8h*)(ap + 16);   // K = ks*32 + hi*8 + 16..23
            v16h a = __builtin_shufflevector(alo, ahi,
                     0, 1, 2, 3, 4, 5, 6, 7, 8, 9, 10, 11, 12, 13, 14, 15);
#pragma unroll
            for (int nt = 0; nt < NT4; ++nt) {
                const v16h bm = *(const v16h*)&sB[(((ks << 2) + nt) * 32 + lane) * 16];
                c[nt] = __builtin_amdgcn_wmma_f32_16x16x32_f16(
                    false, a, false, bm, (short)0, c[nt], false, false);
            }
        }

        // ---- phase 3: epilogue. D VGPR i -> (M = i + hi16*8, N = nt*16 + mrow) ----
#pragma unroll
        for (int nt = 0; nt < NT4; ++nt) {
            float bs = bias[nt * 16 + mrow];
#pragma unroll
            for (int i = 0; i < 8; ++i) {
                int m = i + (hi16 << 3);
                out[(size_t)(base + m) * ODIM + nt * 16 + mrow] = c[nt][i] + bs;
            }
        }

        // ---- phase 4: restore zeros in exactly the slots we wrote ----
        {
            _Float16* ap = aRow + bd + 6 * h0;
            for (int i = 0; i < 50; ++i) ap[6 * i] = (_Float16)0.0f;
        }
    }
}

extern "C" void kernel_launch(void* const* d_in, const int* in_sizes, int n_in,
                              void* d_out, int out_size, void* d_ws, size_t ws_size,
                              hipStream_t stream) {
    const float* t    = (const float*)d_in[0];
    const float* f    = (const float*)d_in[1];
    const int*   band = (const int*)d_in[2];
    const int*   mask = (const int*)d_in[3];   // bool in reference; harness int convention
    const float* W    = (const float*)d_in[4];
    const float* b    = (const float*)d_in[5];
    float* out = (float*)d_out;

    size_t shmem = (size_t)(B_HALVES + A_HALVES) * sizeof(_Float16);  // 247808 B < 320 KB
    plotter_emb_wmma<<<dim3(NBLOCKS), dim3(NTHREADS), shmem, stream>>>(
        t, f, band, mask, W, b, out);
}
